// Attention_49108656062840
// MI455X (gfx1250) — matrix operand order repaired and checked by reading, not compiled, not run on hardware
//
#include <hip/hip_runtime.h>
#include <math.h>

// Problem constants (B,S,DM)=(2,2048,2048), H=16, KV=4, HD=128
#define B_    2
#define S_    2048
#define DM_   2048
#define H_    16
#define KV_   4
#define HD_   128
#define G_    (H_ / KV_)
#define THETA_ 10000.0f
#define EPS_   1e-6f

typedef __attribute__((ext_vector_type(16))) __bf16 v16bf;
typedef __attribute__((ext_vector_type(8)))  float  v8f;
typedef unsigned int uint32x4 __attribute__((ext_vector_type(4)));
typedef int          int32x8  __attribute__((ext_vector_type(8)));
typedef int          int32x4  __attribute__((ext_vector_type(4)));

union FragU { v16bf v; unsigned int u[8]; };

// A-matrix (16x32 bf16) fragment load per CDNA5 ISA layout:
// lane row = lane%16; VGPR v holds K pair at ((v>>2)<<4) + (lane>=16 ? 8:0) + (v&3)*2
__device__ __forceinline__ v16bf frag_a_ld(const __bf16* base, int row_stride, int lane) {
  FragU f;
  const unsigned short* p = (const unsigned short*)base + (lane & 15) * row_stride;
  const int half8 = ((lane >> 4) & 1) << 3;
#pragma unroll
  for (int v = 0; v < 8; ++v) {
    int k = ((v >> 2) << 4) + half8 + ((v & 3) << 1);
    f.u[v] = *(const unsigned int*)(p + k);
  }
  return f.v;
}

// B-matrix (32x16 bf16) fragment load: lane col = lane%16; K pair at ((v>>2)<<4) + (lane>=16 ? 8:0) + (v&3)*2.
// `base` points at a [col][K] (K-contiguous per column) LDS tile.
__device__ __forceinline__ v16bf frag_b_ld(const __bf16* base, int col_stride, int lane) {
  FragU f;
  const unsigned short* p = (const unsigned short*)base + (lane & 15) * col_stride;
  const int kb = ((lane >> 4) & 1) << 3;
#pragma unroll
  for (int v = 0; v < 8; ++v) f.u[v] = *(const unsigned int*)(p + ((v >> 2) << 4) + kb + ((v & 3) << 1));
  return f.v;
}

#define WMMA_BF16(a, b, c) \
  __builtin_amdgcn_wmma_f32_16x16x32_bf16(false, (a), false, (b), (short)0, (c), false, false)

// ---- Tensor Data Mover: 2D tile (bf16) global -> LDS, per CDNA5 D# layout ----
// dims/strides in elements (data_size units); data_size = 2 bytes.
__device__ __forceinline__ unsigned lds_off(const void* p) {
  return (unsigned)(unsigned long long)p;   // low 32 bits = LDS byte offset
}

__device__ __forceinline__ void tdm_load_2d(unsigned lds_addr, const void* gaddr,
                                            unsigned tensor_d0, unsigned tensor_d1,
                                            unsigned tile_d0, unsigned tile_d1,
                                            unsigned long long stride0) {
  unsigned long long ga = (unsigned long long)gaddr;
  uint32x4 g0;
  g0[0] = 1u;                                           // count=1 (valid), user mode
  g0[1] = lds_addr;                                     // lds_addr [63:32]
  g0[2] = (unsigned)(ga & 0xFFFFFFFFu);                 // global_addr [95:64]
  g0[3] = (unsigned)((ga >> 32) & 0x1FFFFFFu) | 0x80000000u;  // ga[56:32] | type=2
  int32x8 g1;
  g1[0] = (int)(1u << 16);                              // data_size=1 -> 2 bytes
  g1[1] = (int)((tensor_d0 & 0xFFFFu) << 16);           // tensor_dim0 [79:48]
  g1[2] = (int)((tensor_d0 >> 16) | ((tensor_d1 & 0xFFFFu) << 16)); // dim0|dim1
  g1[3] = (int)((tensor_d1 >> 16) | (tile_d0 << 16));   // dim1 | tile_dim0 [127:112]
  g1[4] = (int)(tile_d1 & 0xFFFFu);                     // tile_dim1; tile_dim2=0
  g1[5] = (int)(stride0 & 0xFFFFFFFFu);                 // tensor_dim0_stride lo
  g1[6] = (int)((stride0 >> 32) & 0xFFFFu);             // stride hi; dim1_stride=0
  g1[7] = 0;
  int32x4 z4 = {0, 0, 0, 0};
  int32x8 z8 = {0, 0, 0, 0, 0, 0, 0, 0};
  // clang-23 / therock form: (g0, g1, g2, g3, g4, cpol)
  __builtin_amdgcn_tensor_load_to_lds(g0, g1, z4, z4, z8, 0);
}

// ---------------------------------------------------------------- fp32->bf16
__global__ void k_f32_to_bf16(const float* __restrict__ in, __bf16* __restrict__ out, int n) {
  int i = (blockIdx.x * blockDim.x + threadIdx.x) * 4;
  if (i + 3 < n) {
    float4 f = *(const float4*)(in + i);
    out[i + 0] = (__bf16)f.x;
    out[i + 1] = (__bf16)f.y;
    out[i + 2] = (__bf16)f.z;
    out[i + 3] = (__bf16)f.w;
  }
}

// ---------------------------- fp32 [R][C] -> bf16 [C][R] (weights pre-pass)
__global__ __launch_bounds__(256)
void k_transpose_f32_bf16(const float* __restrict__ in, __bf16* __restrict__ out,
                          int R, int C) {
  __shared__ float t[32][33];
  const int tx = threadIdx.x & 31, ty = threadIdx.x >> 5;  // 32x8
  const int c0 = blockIdx.x * 32, r0 = blockIdx.y * 32;
#pragma unroll
  for (int i = 0; i < 32; i += 8)
    t[ty + i][tx] = in[(size_t)(r0 + ty + i) * C + c0 + tx];
  __syncthreads();
#pragma unroll
  for (int i = 0; i < 32; i += 8)
    out[(size_t)(c0 + ty + i) * R + r0 + tx] = (__bf16)t[tx][ty + i];
}

// ------------- fp32 V [b][s][kv][hd] -> bf16 Vt [b][kv][hd][s] (pre-pass)
__global__ __launch_bounds__(256)
void k_transpose_v(const float* __restrict__ in, __bf16* __restrict__ out) {
  __shared__ float t[32][33];
  const int tx = threadIdx.x & 31, ty = threadIdx.x >> 5;
  const int s0 = blockIdx.x * 32, d0 = blockIdx.y * 32;
  const int b = blockIdx.z / KV_, kv = blockIdx.z % KV_;
#pragma unroll
  for (int i = 0; i < 32; i += 8)
    t[ty + i][tx] = in[((size_t)(b * S_ + s0 + ty + i) * KV_ + kv) * HD_ + d0 + tx];
  __syncthreads();
#pragma unroll
  for (int i = 0; i < 32; i += 8)
    out[((size_t)(b * KV_ + kv) * HD_ + d0 + ty + i) * S_ + s0 + tx] = (__bf16)t[tx][ty + i];
}

// ------------------------------------------------- bf16 GEMM, fp32 result
// C[M][N] = A[M][K] * Bt[N][K]^T; block = 128x128 tile, 256 threads (8 waves).
// TDM double-buffered tiles; wave grid 4(m) x 2(n): 8 WMMA/K-step per wave.
__global__ __launch_bounds__(256)
void k_gemm_bf16(const __bf16* __restrict__ A, const __bf16* __restrict__ Bt,
                 float* __restrict__ C, int M, int N, int K) {
  __shared__ __align__(16) __bf16 As[2][128 * 32];  // [row][k]   2 x 8 KB
  __shared__ __align__(16) __bf16 Bs[2][128 * 32];  // [col][k]   2 x 8 KB
  const int tid  = threadIdx.x;
  const int lane = tid & 31;
  const int wave = tid >> 5;
  const int wm   = wave >> 1;          // 0..3  (32-row strip)
  const int wn   = wave & 1;           // 0..1  (64-col strip)
  const int n0 = blockIdx.x * 128;
  const int m0 = blockIdx.y * 128;
  const bool issuer = (wave == 0);
  const v8f vzero = {0.f, 0.f, 0.f, 0.f, 0.f, 0.f, 0.f, 0.f};

  v8f acc[2][4];
#pragma unroll
  for (int mi = 0; mi < 2; ++mi)
#pragma unroll
    for (int nt = 0; nt < 4; ++nt) acc[mi][nt] = vzero;

  if (issuer) {                                       // prologue: buffer 0
    tdm_load_2d(lds_off(&As[0][0]), A + (size_t)m0 * K, K, M, 32, 128, (unsigned long long)K);
    tdm_load_2d(lds_off(&Bs[0][0]), Bt + (size_t)n0 * K, K, N, 32, 128, (unsigned long long)K);
    __builtin_amdgcn_s_wait_tensorcnt(0);
  }
  __syncthreads();

  int buf = 0;
  for (int k0 = 0; k0 < K; k0 += 32) {
    if (issuer && k0 + 32 < K) {                      // async-load next tile
      tdm_load_2d(lds_off(&As[buf ^ 1][0]), A + (size_t)m0 * K + k0 + 32,
                  K, M, 32, 128, (unsigned long long)K);
      tdm_load_2d(lds_off(&Bs[buf ^ 1][0]), Bt + (size_t)n0 * K + k0 + 32,
                  K, N, 32, 128, (unsigned long long)K);
    }
    if (k0 + 64 < K)                                  // warm L2 two steps ahead
      __builtin_prefetch(A + (size_t)(m0 + (tid & 127)) * K + k0 + 64, 0, 1);

    v16bf a0 = frag_a_ld(As[buf] + (wm * 32) * 32, 32, lane);
    v16bf a1 = frag_a_ld(As[buf] + (wm * 32 + 16) * 32, 32, lane);
#pragma unroll
    for (int nt = 0; nt < 4; ++nt) {
      v16bf b = frag_b_ld(Bs[buf] + (wn * 64 + nt * 16) * 32, 32, lane);
      acc[0][nt] = WMMA_BF16(a0, b, acc[0][nt]);
      acc[1][nt] = WMMA_BF16(a1, b, acc[1][nt]);
    }
    if (issuer) __builtin_amdgcn_s_wait_tensorcnt(0);
    __syncthreads();
    buf ^= 1;
  }

  const int half = (lane >> 4) & 1;
  const int cn   = lane & 15;
#pragma unroll
  for (int mi = 0; mi < 2; ++mi)
#pragma unroll
    for (int nt = 0; nt < 4; ++nt)
#pragma unroll
      for (int r = 0; r < 8; ++r) {
        int row = m0 + wm * 32 + mi * 16 + r + 8 * half;
        C[(size_t)row * N + n0 + wn * 64 + nt * 16 + cn] = acc[mi][nt][r];
      }
}

// ---------------------------------------- RMSNorm + RoPE (fp32 in, bf16 out)
// One wave per (b,s,head) row of HD=128. outscale folds in HD^-0.5 for Q.
__global__ __launch_bounds__(128)
void k_norm_rope(const float* __restrict__ in, const float* __restrict__ scale,
                 __bf16* __restrict__ out, int nheads, float outscale) {
  const int lane = threadIdx.x & 31;
  const int wave = threadIdx.x >> 5;
  const int row  = blockIdx.x * 4 + wave;
  const int s    = (row / nheads) % S_;
  const float* x = in + (size_t)row * HD_;
  __bf16*      o = out + (size_t)row * HD_;

  const int i0 = lane * 2;
  float x1a = x[i0], x1b = x[i0 + 1], x2a = x[i0 + 64], x2b = x[i0 + 65];
  float ss = x1a * x1a + x1b * x1b + x2a * x2a + x2b * x2b;
#pragma unroll
  for (int m = 1; m < 32; m <<= 1) ss += __shfl_xor(ss, m, 32);
  float rn = rsqrtf(ss * (1.0f / HD_) + EPS_);

  float n1a = x1a * rn * scale[i0];
  float n1b = x1b * rn * scale[i0 + 1];
  float n2a = x2a * rn * scale[i0 + 64];
  float n2b = x2b * rn * scale[i0 + 65];

  float pos = (float)s;
  float aa = pos / __powf(THETA_, (2.0f * (float)i0) / (float)HD_);
  float ab = pos / __powf(THETA_, (2.0f * (float)(i0 + 1)) / (float)HD_);
  float sa = __sinf(aa), ca = __cosf(aa);
  float sb = __sinf(ab), cb = __cosf(ab);

  o[i0]      = (__bf16)((n1a * ca - n2a * sa) * outscale);
  o[i0 + 1]  = (__bf16)((n1b * cb - n2b * sb) * outscale);
  o[i0 + 64] = (__bf16)((n2a * ca + n1a * sa) * outscale);
  o[i0 + 65] = (__bf16)((n2b * cb + n1b * sb) * outscale);
}

// --------------------------------------------- flash attention (causal, GQA)
// Block: 256 threads (8 waves) handles 128 queries of one (b, h).
// All tiles DMA'd by the Tensor Data Mover, double-buffered K/V.
// LDS = 112 KB (CDNA5 320 KB workgroup budget).
__global__ __launch_bounds__(256)
void k_flash_attn(const __bf16* __restrict__ Q, const __bf16* __restrict__ Kb,
                  const __bf16* __restrict__ Vg, __bf16* __restrict__ Ob) {
  __shared__ __align__(16) __bf16 Qs[128 * HD_];       // [q][hd]        32 KB
  __shared__ __align__(16) __bf16 Ks[2][64 * HD_];     // [key][hd]    2x16 KB
  __shared__ __align__(16) __bf16 Vt[2][HD_ * 64];     // [hd][key]    2x16 KB
  __shared__ __align__(16) __bf16 Ps[8][16 * 64];      // per-wave P     16 KB

  const int tid  = threadIdx.x;
  const int lane = tid & 31;
  const int wave = tid >> 5;
  const int q0  = blockIdx.x * 128;
  const int h   = blockIdx.y;
  const int b   = blockIdx.z;
  const int kvh = h / G_;
  const int half = (lane >> 4) & 1;
  const int cn   = lane & 15;
  const bool issuer = (wave == 0);
  const v8f vzero = {0.f, 0.f, 0.f, 0.f, 0.f, 0.f, 0.f, 0.f};

  const __bf16* kbase = Kb + ((size_t)b * S_ * KV_ + kvh) * HD_;      // + key*KV_*HD_
  const __bf16* vbase = Vg + ((size_t)(b * KV_ + kvh) * HD_) * S_;    // [hd][s]

  if (issuer) {                                        // prologue: Q + K0/V0
    tdm_load_2d(lds_off(Qs), Q + (((size_t)b * S_ + q0) * H_ + h) * HD_,
                HD_, S_, 128, 128, (unsigned long long)(H_ * HD_));
    tdm_load_2d(lds_off(&Ks[0][0]), kbase, HD_, S_, 128, 64,
                (unsigned long long)(KV_ * HD_));
    tdm_load_2d(lds_off(&Vt[0][0]), vbase, S_, HD_, 64, 128,
                (unsigned long long)S_);
    __builtin_amdgcn_s_wait_tensorcnt(0);
  }
  __syncthreads();

  v8f   acc_o[8];
  float mrow[8], lrow[8];
#pragma unroll
  for (int t = 0; t < 8; ++t) { acc_o[t] = vzero; mrow[t] = -3.0e38f; lrow[t] = 0.0f; }

  const int nkt = q0 / 64 + 2;                         // causal: keys <= q0+127
  int buf = 0;
  for (int kt = 0; kt < nkt; ++kt) {
    const int k0 = kt * 64;
    if (issuer && kt + 1 < nkt) {                      // async-load next K/V tile
      tdm_load_2d(lds_off(&Ks[buf ^ 1][0]), kbase + (size_t)(k0 + 64) * KV_ * HD_,
                  HD_, S_, 128, 64, (unsigned long long)(KV_ * HD_));
      tdm_load_2d(lds_off(&Vt[buf ^ 1][0]), vbase + k0 + 64,
                  S_, HD_, 64, 128, (unsigned long long)S_);
    }

    v8f sc[4];                                         // S = Q K^T  (16x64/wave)
#pragma unroll
    for (int t = 0; t < 4; ++t) sc[t] = vzero;
#pragma unroll
    for (int kk = 0; kk < 4; ++kk) {
      v16bf a = frag_a_ld(Qs + wave * 16 * HD_ + kk * 32, HD_, lane);
#pragma unroll
      for (int nt = 0; nt < 4; ++nt) {
        v16bf bm = frag_b_ld(Ks[buf] + nt * 16 * HD_ + kk * 32, HD_, lane);
        sc[nt] = WMMA_BF16(a, bm, sc[nt]);
      }
    }

    float pv[4][8];                                    // causal + online softmax
#pragma unroll
    for (int r = 0; r < 8; ++r) {
      int grow = q0 + wave * 16 + r + 8 * half;
      float mx = -3.0e38f;
#pragma unroll
      for (int nt = 0; nt < 4; ++nt) {
        int gcol = k0 + nt * 16 + cn;
        float v = (gcol <= grow) ? sc[nt][r] : -3.0e38f;
        pv[nt][r] = v;
        mx = fmaxf(mx, v);
      }
#pragma unroll
      for (int m = 1; m < 16; m <<= 1) mx = fmaxf(mx, __shfl_xor(mx, m, 32));
      float mnew = fmaxf(mrow[r], mx);
      float corr = __expf(mrow[r] - mnew);
      mrow[r] = mnew;
      float rsum = 0.0f;
#pragma unroll
      for (int nt = 0; nt < 4; ++nt) {
        float p = __expf(pv[nt][r] - mnew);
        pv[nt][r] = p;
        rsum += p;
      }
#pragma unroll
      for (int m = 1; m < 16; m <<= 1) rsum += __shfl_xor(rsum, m, 32);
      lrow[r] = lrow[r] * corr + rsum;
#pragma unroll
      for (int nt = 0; nt < 8; ++nt) acc_o[nt][r] *= corr;
    }

#pragma unroll
    for (int r = 0; r < 8; ++r)                        // C-layout -> A-layout
#pragma unroll
      for (int nt = 0; nt < 4; ++nt)
        Ps[wave][(r + 8 * half) * 64 + nt * 16 + cn] = (__bf16)pv[nt][r];

#pragma unroll
    for (int kk = 0; kk < 2; ++kk) {                   // O += P @ V
      v16bf a = frag_a_ld(Ps[wave] + kk * 32, 64, lane);
#pragma unroll
      for (int nt = 0; nt < 8; ++nt) {
        v16bf bm = frag_b_ld(Vt[buf] + nt * 16 * 64 + kk * 32, 64, lane);
        acc_o[nt] = WMMA_BF16(a, bm, acc_o[nt]);
      }
    }

    if (issuer) __builtin_amdgcn_s_wait_tensorcnt(0);
    __syncthreads();
    buf ^= 1;
  }

#pragma unroll
  for (int r = 0; r < 8; ++r) {                        // normalize + write att
    float inv = 1.0f / lrow[r];
    int grow = q0 + wave * 16 + r + 8 * half;
#pragma unroll
    for (int nt = 0; nt < 8; ++nt)
      Ob[(((size_t)b * S_ + grow) * H_ + h) * HD_ + nt * 16 + cn] =
          (__bf16)(acc_o[nt][r] * inv);
  }
}

// --------------------------------------------------------------- launcher
extern "C" void kernel_launch(void* const* d_in, const int* in_sizes, int n_in,
                              void* d_out, int out_size, void* d_ws, size_t ws_size,
                              hipStream_t stream) {
  (void)in_sizes; (void)n_in; (void)out_size; (void)ws_size;
  const float* x       = (const float*)d_in[0];
  const float* wq      = (const float*)d_in[1];
  const float* wk      = (const float*)d_in[2];
  const float* wv      = (const float*)d_in[3];
  const float* wo      = (const float*)d_in[4];
  const float* q_scale = (const float*)d_in[5];
  const float* k_scale = (const float*)d_in[6];
  float* out = (float*)d_out;

  const size_t N_X   = (size_t)B_ * S_ * DM_;        // 8388608
  const size_t N_WQ  = (size_t)DM_ * H_ * HD_;       // 4194304
  const size_t N_WKV = (size_t)DM_ * KV_ * HD_;      // 1048576
  const size_t N_Q   = (size_t)B_ * S_ * H_ * HD_;   // 8388608
  const size_t N_KV  = (size_t)B_ * S_ * KV_ * HD_;  // 2097152

  char* w = (char*)d_ws;
  size_t off = 0;
  auto take = [&](size_t bytes) { char* p = w + off; off += (bytes + 255) & ~(size_t)255; return p; };
  __bf16* xb   = (__bf16*)take(N_X  * 2);   // x           [M][K]
  __bf16* wqt  = (__bf16*)take(N_WQ * 2);   // wq^T        [N][K]
  __bf16* wkt  = (__bf16*)take(N_WKV * 2);  // wk^T        [N][K]
  __bf16* wvt  = (__bf16*)take(N_WKV * 2);  // wv^T        [N][K]
  __bf16* wot  = (__bf16*)take(N_WQ * 2);   // wo^T        [N][K]
  __bf16* qb   = (__bf16*)take(N_Q  * 2);   // q (normed)  [b][s][h][hd]
  __bf16* kb   = (__bf16*)take(N_KV * 2);   // k (normed)  [b][s][kv][hd]
  __bf16* vtb  = (__bf16*)take(N_KV * 2);   // v           [b][kv][hd][s]
  __bf16* attb = (__bf16*)take(N_Q  * 2);   // attention   [b][s][h][hd]
  float*  qf   = (float*)take(N_Q  * 4);
  float*  kf   = (float*)take(N_KV * 4);
  float*  vf   = (float*)take(N_KV * 4);

  // prep: x -> bf16; weights -> transposed bf16 [N][K]
  k_f32_to_bf16<<<(int)((N_X / 4 + 255) / 256), 256, 0, stream>>>(x, xb, (int)N_X);
  k_transpose_f32_bf16<<<dim3((H_ * HD_) / 32, DM_ / 32), 256, 0, stream>>>(wq, wqt, DM_, H_ * HD_);
  k_transpose_f32_bf16<<<dim3((KV_ * HD_) / 32, DM_ / 32), 256, 0, stream>>>(wk, wkt, DM_, KV_ * HD_);
  k_transpose_f32_bf16<<<dim3((KV_ * HD_) / 32, DM_ / 32), 256, 0, stream>>>(wv, wvt, DM_, KV_ * HD_);
  k_transpose_f32_bf16<<<dim3(DM_ / 32, (H_ * HD_) / 32), 256, 0, stream>>>(wo, wot, H_ * HD_, DM_);

  const int M = B_ * S_;                             // 4096
  k_gemm_bf16<<<dim3((H_ * HD_) / 128, M / 128), 256, 0, stream>>>(xb, wqt, qf, M, H_ * HD_, DM_);
  k_gemm_bf16<<<dim3((KV_ * HD_) / 128, M / 128), 256, 0, stream>>>(xb, wkt, kf, M, KV_ * HD_, DM_);
  k_gemm_bf16<<<dim3((KV_ * HD_) / 128, M / 128), 256, 0, stream>>>(xb, wvt, vf, M, KV_ * HD_, DM_);

  k_norm_rope<<<(B_ * S_ * H_) / 4, 128, 0, stream>>>(qf, q_scale, qb, H_, 0.08838834764831845f);
  k_norm_rope<<<(B_ * S_ * KV_) / 4, 128, 0, stream>>>(kf, k_scale, kb, KV_, 1.0f);
  k_transpose_v<<<dim3(S_ / 32, HD_ / 32, B_ * KV_), 256, 0, stream>>>(vf, vtb);

  k_flash_attn<<<dim3(S_ / 128, H_, B_), 256, 0, stream>>>(qb, kb, vtb, attb);

  k_gemm_bf16<<<dim3(DM_ / 128, M / 128), 256, 0, stream>>>(attb, wot, out, M, DM_, DM_);
}
